// Foveal_4423816315042
// MI455X (gfx1250) — matrix operand order repaired and checked by reading, not compiled, not run on hardware
//
#include <hip/hip_runtime.h>
#include <hip/hip_bf16.h>
#include <math.h>

// ---------------- problem constants ----------------
#define CB    64      // batch
#define CN    4096    // sequence length of x
#define CD    1024    // model dim
#define CNH   16      // heads
#define CDH   64      // head dim
#define CWS   256     // window size
#define CA    256     // acc length
#define CL    513     // tokens = 1 + CA + CWS
#define CLP   528     // padded to multiple of 16 (33 tiles)
#define CFF   4096    // ffn hidden
#define CEPS  1e-5f

typedef __bf16 bf16;
typedef __attribute__((ext_vector_type(8)))  bf16  v8bf;
typedef __attribute__((ext_vector_type(16))) bf16  v16bf;
typedef __attribute__((ext_vector_type(8)))  float v8f;

// ---------------- K0: in_w rows [1024,3072) -> bf16 W_kv ----------------
__global__ void fov_conv_wkv(const float* __restrict__ in_w, bf16* __restrict__ wkv) {
    const size_t total = (size_t)2 * CD * CD;            // 2048*1024
    for (size_t i = (size_t)blockIdx.x * blockDim.x + threadIdx.x; i < total;
         i += (size_t)gridDim.x * blockDim.x)
        wkv[i] = (bf16)in_w[(size_t)CD * CD + i];
}

// ---------------- K1: gather ctx row + LayerNorm -> y(bf16); emit out2 ----------------
__global__ void fov_gather_ln(const float* __restrict__ x, const int* __restrict__ fp,
                              const float* __restrict__ state, const float* __restrict__ acc,
                              const float* __restrict__ n1g, const float* __restrict__ n1b,
                              bf16* __restrict__ y, float* __restrict__ out2) {
    const int token = blockIdx.x % CLP;
    const int b     = blockIdx.x / CLP;
    const int tid   = threadIdx.x;
    __shared__ float red[256];

    bf16* yr = y + ((size_t)b * CLP + token) * CD;
    if (token >= CL) {                                   // zero padded rows
        for (int i = tid; i < CD; i += 256) yr[i] = (bf16)0.0f;
        return;
    }
    // window start (matches reference _extract)
    const int c  = fp[b];
    const int h  = CWS / 2;
    const int s0 = max(0, c - h);
    const int e0 = min(CN, c + h);
    const bool need = (e0 - s0) < CWS;
    const int s_adj = (s0 > 0) ? max(0, e0 - CWS) : 0;
    const int s  = need ? s_adj : s0;
    const int e  = need ? min(CN, s + CWS) : e0;

    float v[4];
    #pragma unroll
    for (int j = 0; j < 4; ++j) {
        const int d = tid + 256 * j;
        float val;
        if (token == 0) {
            val = state[(size_t)b * CD + d];
        } else if (token <= CA) {
            val = acc[((size_t)b * CA + (token - 1)) * CD + d];
        } else {
            const int idx = s + (token - 1 - CA);
            val = (idx < e) ? x[((size_t)b * CN + idx) * CD + d] : 0.0f;
        }
        v[j] = val;
        if (token >= 1)                                  // out2 = concat(acc, win)
            out2[((size_t)b * (CA + CWS) + (token - 1)) * CD + d] = val;
    }
    // mean
    float ls = v[0] + v[1] + v[2] + v[3];
    red[tid] = ls; __syncthreads();
    for (int o = 128; o > 0; o >>= 1) { if (tid < o) red[tid] += red[tid + o]; __syncthreads(); }
    const float m = red[0] * (1.0f / CD); __syncthreads();
    // variance
    float lv = 0.0f;
    #pragma unroll
    for (int j = 0; j < 4; ++j) { const float dzz = v[j] - m; lv += dzz * dzz; }
    red[tid] = lv; __syncthreads();
    for (int o = 128; o > 0; o >>= 1) { if (tid < o) red[tid] += red[tid + o]; __syncthreads(); }
    const float rs = rsqrtf(red[0] * (1.0f / CD) + CEPS);
    #pragma unroll
    for (int j = 0; j < 4; ++j) {
        const int d = tid + 256 * j;
        yr[d] = (bf16)((v[j] - m) * rs * n1g[d] + n1b[d]);
    }
}

// ---------------- K2: WMMA bf16 GEMM  kv = y @ Wkv^T + b_kv ----------------
// grid = (2048/512, 33, B). 256 threads = 8 waves; each wave computes a 16x64
// strip (4 accumulators) so the A fragment is loaded once per k-step and
// reused across 4 v_wmma ops (4x less A traffic, 13 FLOP/byte of tile traffic).
__global__ void fov_kv_gemm(const bf16* __restrict__ y, const bf16* __restrict__ wkv,
                            const float* __restrict__ in_b, bf16* __restrict__ kv) {
    const int lane    = threadIdx.x & 31;
    const int wave    = threadIdx.x >> 5;
    const int colbase = (blockIdx.x * 8 + wave) * 64;    // 0..1984 step 64
    const int tileM   = blockIdx.y;                      // 0..32
    const int b       = blockIdx.z;
    const int mrow    = lane & 15;
    const int kh      = lane >> 4;

    const bf16* arow = y + ((size_t)b * CLP + tileM * 16 + mrow) * CD;
    const bf16* brow0 = wkv + ((size_t)(colbase +  0 + mrow)) * CD;
    const bf16* brow1 = wkv + ((size_t)(colbase + 16 + mrow)) * CD;
    const bf16* brow2 = wkv + ((size_t)(colbase + 32 + mrow)) * CD;
    const bf16* brow3 = wkv + ((size_t)(colbase + 48 + mrow)) * CD;

    v8f c0 = {}, c1 = {}, c2 = {}, c3 = {};
    #pragma unroll 2
    for (int kb = 0; kb < CD; kb += 32) {
        // A fragment: lane holds 16 contiguous K values (16*kh .. 16*kh+15), matching B
        v8bf a0 = *(const v8bf*)(arow + kb + 16 * kh);
        v8bf a1 = *(const v8bf*)(arow + kb + 16 * kh + 8);
        v16bf av = __builtin_shufflevector(a0, a1, 0,1,2,3,4,5,6,7,8,9,10,11,12,13,14,15);
        // B fragments: lane n holds 16 contiguous K values (16*kh .. 16*kh+15)
        v16bf bv0 = *(const v16bf*)(brow0 + kb + 16 * kh);
        v16bf bv1 = *(const v16bf*)(brow1 + kb + 16 * kh);
        v16bf bv2 = *(const v16bf*)(brow2 + kb + 16 * kh);
        v16bf bv3 = *(const v16bf*)(brow3 + kb + 16 * kh);
        c0 = __builtin_amdgcn_wmma_f32_16x16x32_bf16(false, av, false, bv0, (short)0, c0, false, false);
        c1 = __builtin_amdgcn_wmma_f32_16x16x32_bf16(false, av, false, bv1, (short)0, c1, false, false);
        c2 = __builtin_amdgcn_wmma_f32_16x16x32_bf16(false, av, false, bv2, (short)0, c2, false, false);
        c3 = __builtin_amdgcn_wmma_f32_16x16x32_bf16(false, av, false, bv3, (short)0, c3, false, false);
    }
    // C/D layout: c[r] -> M = r + 8*kh, N = lane&15
    v8f cs[4] = {c0, c1, c2, c3};
    #pragma unroll
    for (int j = 0; j < 4; ++j) {
        const int col   = colbase + j * 16 + mrow;
        const float bias = in_b[CD + col];
        #pragma unroll
        for (int r = 0; r < 8; ++r) {
            const int mm = tileM * 16 + r + 8 * kh;
            kv[((size_t)b * CLP + mm) * (2 * CD) + col] = (bf16)(cs[j][r] + bias);
        }
    }
}

// ---------------- K3: q0 GEMV (row 0 of y only) ----------------
__global__ void fov_q0(const bf16* __restrict__ y, const float* __restrict__ in_w,
                       const float* __restrict__ in_b, float* __restrict__ q0) {
    const int gw   = (blockIdx.x * blockDim.x + threadIdx.x) >> 5;
    const int lane = threadIdx.x & 31;
    const int d = gw % CD, b = gw / CD;
    const bf16*  yr = y    + (size_t)b * CLP * CD;       // token 0
    const float* wr = in_w + (size_t)d * CD;
    float s = 0.0f;
    for (int k = lane; k < CD; k += 32) s += (float)yr[k] * wr[k];
    for (int m = 16; m; m >>= 1) s += __shfl_xor(s, m, 32);
    if (lane == 0) q0[(size_t)b * CD + d] = s + in_b[d];
}

// ---------------- K4: single-query attention per (b, head) ----------------
__global__ void fov_attn(const bf16* __restrict__ kv, const float* __restrict__ q0,
                         float* __restrict__ o0) {
    const int hh = blockIdx.x % CNH;
    const int b  = blockIdx.x / CNH;
    const int tid = threadIdx.x;
    __shared__ float sc[CL];
    __shared__ float red[256];
    __shared__ float qh[CDH];
    const bf16* kvb = kv + (size_t)b * CLP * (2 * CD);
    if (tid < CDH) qh[tid] = q0[(size_t)b * CD + hh * CDH + tid];
    __syncthreads();

    for (int t = tid; t < CL; t += 256) {
        const bf16* kr = kvb + (size_t)t * (2 * CD) + hh * CDH;
        float s = 0.0f;
        #pragma unroll
        for (int dd = 0; dd < CDH; dd += 8) {
            v8bf kk = *(const v8bf*)(kr + dd);
            #pragma unroll
            for (int j = 0; j < 8; ++j) s += qh[dd + j] * (float)kk[j];
        }
        sc[t] = s * 0.125f;                              // 1/sqrt(64)
    }
    __syncthreads();
    float lm = -1e30f;
    for (int t = tid; t < CL; t += 256) lm = fmaxf(lm, sc[t]);
    red[tid] = lm; __syncthreads();
    for (int o = 128; o > 0; o >>= 1) { if (tid < o) red[tid] = fmaxf(red[tid], red[tid + o]); __syncthreads(); }
    const float mx = red[0]; __syncthreads();
    float lsum = 0.0f;
    for (int t = tid; t < CL; t += 256) { const float p = expf(sc[t] - mx); sc[t] = p; lsum += p; }
    red[tid] = lsum; __syncthreads();
    for (int o = 128; o > 0; o >>= 1) { if (tid < o) red[tid] += red[tid + o]; __syncthreads(); }
    const float inv = 1.0f / red[0]; __syncthreads();
    // weighted V: 4 groups of 64 threads each cover a quarter of the keys
    const int dq = tid & 63;
    const int g  = tid >> 6;                             // 0..3
    float a = 0.0f;
    for (int t = g; t < CL; t += 4)
        a += sc[t] * (float)kvb[(size_t)t * (2 * CD) + CD + hh * CDH + dq];
    red[tid] = a; __syncthreads();
    if (tid < CDH) {
        const float tot = red[tid] + red[tid + 64] + red[tid + 128] + red[tid + 192];
        o0[(size_t)b * CD + hh * CDH + tid] = tot * inv;
    }
}

// ---------------- K5: out-proj + residual (ctx row0 = state) ----------------
__global__ void fov_outproj(const float* __restrict__ o0, const float* __restrict__ out_w,
                            const float* __restrict__ out_b, const float* __restrict__ state,
                            float* __restrict__ clpre) {
    const int gw = (blockIdx.x * blockDim.x + threadIdx.x) >> 5;
    const int lane = threadIdx.x & 31;
    const int d = gw % CD, b = gw / CD;
    const float* ov = o0    + (size_t)b * CD;
    const float* wr = out_w + (size_t)d * CD;
    float s = 0.0f;
    for (int k = lane; k < CD; k += 32) s += ov[k] * wr[k];
    for (int m = 16; m; m >>= 1) s += __shfl_xor(s, m, 32);
    if (lane == 0) clpre[(size_t)b * CD + d] = state[(size_t)b * CD + d] + s + out_b[d];
}

// ---------------- K6: LayerNorm2 on cl ----------------
__global__ void fov_ln2(const float* __restrict__ clpre, const float* __restrict__ n2g,
                        const float* __restrict__ n2b, float* __restrict__ z) {
    const int b = blockIdx.x, tid = threadIdx.x;
    __shared__ float red[256];
    float v[4];
    #pragma unroll
    for (int j = 0; j < 4; ++j) v[j] = clpre[(size_t)b * CD + tid + 256 * j];
    float ls = v[0] + v[1] + v[2] + v[3];
    red[tid] = ls; __syncthreads();
    for (int o = 128; o > 0; o >>= 1) { if (tid < o) red[tid] += red[tid + o]; __syncthreads(); }
    const float m = red[0] * (1.0f / CD); __syncthreads();
    float lv = 0.0f;
    #pragma unroll
    for (int j = 0; j < 4; ++j) { const float dz = v[j] - m; lv += dz * dz; }
    red[tid] = lv; __syncthreads();
    for (int o = 128; o > 0; o >>= 1) { if (tid < o) red[tid] += red[tid + o]; __syncthreads(); }
    const float rs = rsqrtf(red[0] * (1.0f / CD) + CEPS);
    #pragma unroll
    for (int j = 0; j < 4; ++j) {
        const int d = tid + 256 * j;
        z[(size_t)b * CD + d] = (v[j] - m) * rs * n2g[d] + n2b[d];
    }
}

// ---------------- K7: FF1 + exact GELU ----------------
__global__ void fov_ff1(const float* __restrict__ z, const float* __restrict__ w1,
                        const float* __restrict__ b1, float* __restrict__ hdn) {
    const int gw = (blockIdx.x * blockDim.x + threadIdx.x) >> 5;
    const int lane = threadIdx.x & 31;
    const int i = gw % CFF, b = gw / CFF;
    const float* zr = z  + (size_t)b * CD;
    const float* wr = w1 + (size_t)i * CD;
    float s = 0.0f;
    for (int k = lane; k < CD; k += 32) s += zr[k] * wr[k];
    for (int m = 16; m; m >>= 1) s += __shfl_xor(s, m, 32);
    if (lane == 0) {
        const float hv = s + b1[i];
        hdn[(size_t)b * CFF + i] = 0.5f * hv * (1.0f + erff(hv * 0.70710678118654752f));
    }
}

// ---------------- K8: FF2 + residual -> cl output ----------------
__global__ void fov_ff2(const float* __restrict__ hdn, const float* __restrict__ w2,
                        const float* __restrict__ b2, const float* __restrict__ clpre,
                        float* __restrict__ out1) {
    const int gw = (blockIdx.x * blockDim.x + threadIdx.x) >> 5;
    const int lane = threadIdx.x & 31;
    const int d = gw % CD, b = gw / CD;
    const float* hr = hdn + (size_t)b * CFF;
    const float* wr = w2  + (size_t)d * CFF;
    float s = 0.0f;
    for (int k = lane; k < CFF; k += 32) s += hr[k] * wr[k];
    for (int m = 16; m; m >>= 1) s += __shfl_xor(s, m, 32);
    if (lane == 0) out1[(size_t)b * CD + d] = clpre[(size_t)b * CD + d] + s + b2[d];
}

// ---------------- launcher ----------------
extern "C" void kernel_launch(void* const* d_in, const int* in_sizes, int n_in,
                              void* d_out, int out_size, void* d_ws, size_t ws_size,
                              hipStream_t stream) {
    const float* x     = (const float*)d_in[0];
    const int*   fp    = (const int*)  d_in[1];
    const float* state = (const float*)d_in[2];
    const float* acc   = (const float*)d_in[3];
    const float* n1g   = (const float*)d_in[4];
    const float* n1b   = (const float*)d_in[5];
    const float* n2g   = (const float*)d_in[6];
    const float* n2b   = (const float*)d_in[7];
    const float* in_w  = (const float*)d_in[8];
    const float* in_b  = (const float*)d_in[9];
    const float* out_w = (const float*)d_in[10];
    const float* out_b = (const float*)d_in[11];
    const float* ff1w  = (const float*)d_in[12];
    const float* ff1b  = (const float*)d_in[13];
    const float* ff2w  = (const float*)d_in[14];
    const float* ff2b  = (const float*)d_in[15];

    float* out1 = (float*)d_out;                          // cl: B*D
    float* out2 = out1 + (size_t)CB * CD;                 // concat(acc,win): B*512*D

    // workspace carve-up (256B aligned)
    char* ws = (char*)d_ws;
    size_t off = 0;
    auto take = [&](size_t bytes) { void* p = ws + off; off = (off + bytes + 255) & ~(size_t)255; return p; };
    bf16*  y     = (bf16*) take((size_t)CB * CLP * CD * sizeof(bf16));        // ~69 MB
    bf16*  wkv   = (bf16*) take((size_t)2 * CD * CD * sizeof(bf16));          //  4 MB
    bf16*  kv    = (bf16*) take((size_t)CB * CLP * 2 * CD * sizeof(bf16));    // ~138 MB
    float* q0    = (float*)take((size_t)CB * CD * sizeof(float));
    float* o0    = (float*)take((size_t)CB * CD * sizeof(float));
    float* clpre = (float*)take((size_t)CB * CD * sizeof(float));
    float* zbuf  = (float*)take((size_t)CB * CD * sizeof(float));
    float* hdn   = (float*)take((size_t)CB * CFF * sizeof(float));
    (void)ws_size; (void)in_sizes; (void)n_in; (void)out_size;

    fov_conv_wkv <<<2048, 256, 0, stream>>>(in_w, wkv);
    fov_gather_ln<<<CB * CLP, 256, 0, stream>>>(x, fp, state, acc, n1g, n1b, y, out2);
    fov_kv_gemm  <<<dim3(4, CLP / 16, CB), 256, 0, stream>>>(y, wkv, in_b, kv);
    fov_q0       <<<(CB * CD) / 8, 256, 0, stream>>>(y, in_w, in_b, q0);
    fov_attn     <<<CB * CNH, 256, 0, stream>>>(kv, q0, o0);
    fov_outproj  <<<(CB * CD) / 8, 256, 0, stream>>>(o0, out_w, out_b, state, clpre);
    fov_ln2      <<<CB, 256, 0, stream>>>(clpre, n2g, n2b, zbuf);
    fov_ff1      <<<(CB * CFF) / 8, 256, 0, stream>>>(zbuf, ff1w, ff1b, hdn);
    fov_ff2      <<<(CB * CD) / 8, 256, 0, stream>>>(hdn, ff2w, ff2b, clpre, out1);
}
